// MemoryEfficientAttention_23235773072080
// MI455X (gfx1250) — matrix operand order repaired and checked by reading, not compiled, not run on hardware
//
#include <hip/hip_runtime.h>
#include <hip/hip_bf16.h>
#include <math.h>

typedef __bf16 bf16;
typedef __attribute__((ext_vector_type(16))) __bf16 v16bf;
typedef __attribute__((ext_vector_type(4)))  __bf16 v4bf;
typedef __attribute__((ext_vector_type(2)))  __bf16 v2bf;
typedef __attribute__((ext_vector_type(8)))  float   v8f;
typedef __attribute__((ext_vector_type(4)))  float   v4f;

#define EMBED 2048
#define SEQ   4096
#define HEADS 16
#define DH    128

// ---------------------------------------------------------------------------
// WMMA fragment loaders for v_wmma_f32_16x16x32_bf16 (wave32), per ISA 7.12.2.
// Both collapse to two ds_load_b128 per fragment (verified in round-1 disasm).
// ---------------------------------------------------------------------------
__device__ __forceinline__ v16bf load_frag_a(const bf16* base, int ld) {
  const int lane = threadIdx.x & 31;
  const int m  = lane & 15;
  const int kg = (lane >> 4) * 8;
  v16bf f;
#pragma unroll
  for (int v = 0; v < 8; ++v) {
    const int kk = ((v & 4) ? 16 : 0) + kg + 2 * (v & 3);
    v2bf p = *(const v2bf*)(base + m * ld + kk);
    f[2 * v]     = p.x;
    f[2 * v + 1] = p.y;
  }
  return f;
}

// B tile stored in LDS as [n][k], k contiguous, row stride ld.
__device__ __forceinline__ v16bf load_frag_b(const bf16* base, int ld) {
  const int lane = threadIdx.x & 31;
  const int n  = lane & 15;
  const int kg = (lane >> 4) * 8;
  v16bf f;
#pragma unroll
  for (int v = 0; v < 8; ++v) {
    const int kk = ((v & 4) ? 16 : 0) + kg + 2 * (v & 3);
    v2bf p = *(const v2bf*)(base + n * ld + kk);
    f[2 * v]     = p.x;
    f[2 * v + 1] = p.y;
  }
  return f;
}

// truncate a flat pointer to an LDS byte offset (ISA: LDS aperture uses addr[31:0])
__device__ __forceinline__ unsigned lds_off(const void* p) {
  return (unsigned)(unsigned long long)(uintptr_t)p;
}

// ---------------------------------------------------------------------------
// C = X * W^T + bias   (torch Linear).  X: [M,K], W: [N,K], C: [M,N]
// WG tile 128x128, 8 waves, each wave 32x64 (2x4 accumulators).
// Double-buffered LDS: global->VGPR for slab i+1, WMMA on slab i,
// VGPR->LDS into the other buffer, one barrier per K step.
// ---------------------------------------------------------------------------
template<bool IN_BF16, bool OUT_BF16>
__global__ __launch_bounds__(256) void gemm_xwt_wmma(
    const float* __restrict__ Xf, const bf16* __restrict__ Xb,
    const float* __restrict__ W,  const float* __restrict__ bias,
    float* __restrict__ Yf, bf16* __restrict__ Yb,
    int M, int N, int K)
{
  __shared__ bf16 As[2][128 * 36];   // 128 rows x 32 K, +4 pad
  __shared__ bf16 Bs[2][128 * 36];

  const int t    = threadIdx.x;
  const int lane = t & 31;
  const int wave = t >> 5;
  const int wm   = wave & 3;        // 4 waves along M (32 rows each)
  const int wn   = wave >> 2;       // 2 waves along N (64 cols each)
  const size_t m0 = (size_t)blockIdx.y * 128;
  const size_t n0 = (size_t)blockIdx.x * 128;

  v8f zero = {};
  v8f acc[2][4];
#pragma unroll
  for (int i = 0; i < 2; ++i)
#pragma unroll
    for (int j = 0; j < 4; ++j) acc[i][j] = zero;

  // staging registers
  v4f  areg[4], breg[4];
  v4bf abreg[2][2];

  auto gload = [&](int kc) {
#pragma unroll
    for (int i = 0; i < 4; ++i) {
      const int idx = t + 256 * i;
      const int row = idx >> 3;
      const int c4  = (idx & 7) * 4;
      breg[i] = *(const v4f*)(W + (n0 + row) * (size_t)K + kc + c4);
      if (!IN_BF16)
        areg[i] = *(const v4f*)(Xf + (m0 + row) * (size_t)K + kc + c4);
    }
    if (IN_BF16) {
#pragma unroll
      for (int i = 0; i < 2; ++i) {
        const int idx = t + 256 * i;
        const int row = idx >> 2;
        const int c8  = (idx & 3) * 8;
        const bf16* src = Xb + (m0 + row) * (size_t)K + kc + c8;
        abreg[i][0] = *(const v4bf*)(src);
        abreg[i][1] = *(const v4bf*)(src + 4);
      }
    }
  };

  auto lstore = [&](bf16* As_, bf16* Bs_) {
#pragma unroll
    for (int i = 0; i < 4; ++i) {
      const int idx = t + 256 * i;
      const int row = idx >> 3;
      const int c4  = (idx & 7) * 4;
      v4bf b;
      b.x = (bf16)breg[i].x; b.y = (bf16)breg[i].y;
      b.z = (bf16)breg[i].z; b.w = (bf16)breg[i].w;
      *(v4bf*)(Bs_ + row * 36 + c4) = b;
      if (!IN_BF16) {
        v4bf a;
        a.x = (bf16)areg[i].x; a.y = (bf16)areg[i].y;
        a.z = (bf16)areg[i].z; a.w = (bf16)areg[i].w;
        *(v4bf*)(As_ + row * 36 + c4) = a;
      }
    }
    if (IN_BF16) {
#pragma unroll
      for (int i = 0; i < 2; ++i) {
        const int idx = t + 256 * i;
        const int row = idx >> 2;
        const int c8  = (idx & 3) * 8;
        *(v4bf*)(As_ + row * 36 + c8)     = abreg[i][0];
        *(v4bf*)(As_ + row * 36 + c8 + 4) = abreg[i][1];
      }
    }
  };

  gload(0);
  lstore(As[0], Bs[0]);
  __syncthreads();

  int buf = 0;
  for (int kc = 0; kc < K; kc += 32) {
    const bool more = (kc + 32 < K);
    if (more) {
      gload(kc + 32);
      // L2 prefetch hint for slab after next
      if (kc + 64 < K)
        __builtin_prefetch(W + (n0 + (t >> 1)) * (size_t)K + kc + 64 + (t & 1) * 16, 0, 1);
    }

    v16bf afr[2], bfr[4];
#pragma unroll
    for (int i = 0; i < 2; ++i)
      afr[i] = load_frag_a(As[buf] + (wm * 32 + i * 16) * 36, 36);
#pragma unroll
    for (int j = 0; j < 4; ++j)
      bfr[j] = load_frag_b(Bs[buf] + (wn * 64 + j * 16) * 36, 36);
#pragma unroll
    for (int i = 0; i < 2; ++i)
#pragma unroll
      for (int j = 0; j < 4; ++j)
        acc[i][j] = __builtin_amdgcn_wmma_f32_16x16x32_bf16(
            false, afr[i], false, bfr[j], (short)0, acc[i][j], false, false);

    if (more) lstore(As[buf ^ 1], Bs[buf ^ 1]);
    __syncthreads();
    buf ^= 1;
  }

  // ---- epilogue ----
  const int half = lane >> 4;
  const int nI   = lane & 15;
#pragma unroll
  for (int i = 0; i < 2; ++i) {
#pragma unroll
    for (int j = 0; j < 4; ++j) {
      const size_t col = n0 + wn * 64 + j * 16 + nI;
      const float bv = bias[col];
      const size_t mb = m0 + wm * 32 + i * 16 + half * 8;
#pragma unroll
      for (int r = 0; r < 8; ++r) {
        const float val = acc[i][j][r] + bv;
        if (OUT_BF16) Yb[(mb + r) * (size_t)N + col] = (bf16)val;
        else          Yf[(mb + r) * (size_t)N + col] = val;
      }
    }
  }
}

// ---------------------------------------------------------------------------
// Flash attention: one WG (128 threads / 4 waves) per (64-query block, head).
// Q,K,V are bf16 [L][EMBED] (head h at columns h*DH..h*DH+127).
// K tile staged with GLOBAL_LOAD_ASYNC_TO_LDS_B128 (ASYNCcnt-tracked,
// no VGPR round-trip); V tile transposed manually (async can't transpose).
// ---------------------------------------------------------------------------
__global__ __launch_bounds__(128) void flash_attn_wmma(
    const bf16* __restrict__ Q, const bf16* __restrict__ Kh,
    const bf16* __restrict__ V, bf16* __restrict__ O,
    const int* __restrict__ is_causal_p)
{
  __shared__ bf16 Ks[64 * 136];     // [key][dh]  (= B layout for Q*K^T)
  __shared__ bf16 Vs[DH * 68];      // [dh][key]  (= B layout for P*V)
  __shared__ bf16 Ps[4][16 * 72];   // per-wave P tile [q][key]

  const int t    = threadIdx.x;
  const int lane = t & 31;
  const int wave = t >> 5;
  const int half = lane >> 4;
  const int nI   = lane & 15;
  const int qb   = blockIdx.x;
  const int h    = blockIdx.y;
  const bool causal = (is_causal_p[0] != 0);
  const float scale = 0.088388347648318447f;   // 1/sqrt(128)

  // Build Q A-fragments directly from global: each lane needs two contiguous
  // 16B runs per fragment (K kg..kg+7 and K 16+kg..16+kg+7 of its row).
  const int m  = lane & 15;
  const int kg = (lane >> 4) * 8;
  const bf16* qrow = Q + ((size_t)(qb * 64 + wave * 16 + m)) * EMBED + h * DH;
  v16bf aq[4];
#pragma unroll
  for (int d = 0; d < 4; ++d) {
    v4bf a0 = *(const v4bf*)(qrow + d * 32 + kg);
    v4bf a1 = *(const v4bf*)(qrow + d * 32 + kg + 4);
    v4bf a2 = *(const v4bf*)(qrow + d * 32 + kg + 16);
    v4bf a3 = *(const v4bf*)(qrow + d * 32 + kg + 20);
    aq[d][0]  = a0.x; aq[d][1]  = a0.y; aq[d][2]  = a0.z; aq[d][3]  = a0.w;
    aq[d][4]  = a1.x; aq[d][5]  = a1.y; aq[d][6]  = a1.z; aq[d][7]  = a1.w;
    aq[d][8]  = a2.x; aq[d][9]  = a2.y; aq[d][10] = a2.z; aq[d][11] = a2.w;
    aq[d][12] = a3.x; aq[d][13] = a3.y; aq[d][14] = a3.z; aq[d][15] = a3.w;
  }

  v8f zero = {};
  v8f oacc[8];
#pragma unroll
  for (int j = 0; j < 8; ++j) oacc[j] = zero;
  float mrow[8], lrow[8];
#pragma unroll
  for (int r = 0; r < 8; ++r) { mrow[r] = -1.0e30f; lrow[r] = 0.0f; }

  const int kb_last = causal ? qb : (SEQ / 64 - 1);
  for (int kb = 0; kb <= kb_last; ++kb) {
    __syncthreads();   // previous iteration's LDS reads are done

    // ---- stage K block via async copy: 64x128 bf16 = 1024 x 16B chunks ----
#pragma unroll
    for (int i = 0; i < 8; ++i) {
      const int idx = t + 128 * i;
      const int row = idx >> 4;
      const int c8  = (idx & 15) * 8;
      const bf16* ksrc = Kh + ((size_t)(kb * 64 + row)) * EMBED + h * DH + c8;
      const unsigned dst = lds_off(Ks + row * 136 + c8);
      const unsigned long long ga = (unsigned long long)(uintptr_t)ksrc;
      asm volatile("global_load_async_to_lds_b128 %0, %1, off"
                   :: "v"(dst), "v"(ga) : "memory");
    }
    // ---- stage V block transposed ([dh][key]) through VGPRs ----
#pragma unroll
    for (int i = 0; i < 8; ++i) {
      const int idx = t + 128 * i;
      const int row = idx >> 4;
      const int c8  = (idx & 15) * 8;
      const bf16* vsrc = V + ((size_t)(kb * 64 + row)) * EMBED + h * DH + c8;
#pragma unroll
      for (int e = 0; e < 8; ++e) Vs[(c8 + e) * 68 + row] = vsrc[e];
    }
    asm volatile("s_wait_asynccnt 0x0" ::: "memory");
    __syncthreads();

    // ---- S = Q * K^T  (per-wave 16x64) ----
    v8f sacc[4];
#pragma unroll
    for (int j = 0; j < 4; ++j) sacc[j] = zero;
#pragma unroll
    for (int j = 0; j < 4; ++j)
#pragma unroll
      for (int d = 0; d < 4; ++d)
        sacc[j] = __builtin_amdgcn_wmma_f32_16x16x32_bf16(
            false, aq[d], false, load_frag_b(Ks + (j * 16) * 136 + d * 32, 136),
            (short)0, sacc[j], false, false);

    float sv[4][8];
    const bool diag = causal && (kb == qb);
#pragma unroll
    for (int j = 0; j < 4; ++j)
#pragma unroll
      for (int r = 0; r < 8; ++r) {
        float x = sacc[j][r] * scale;
        if (diag) {
          const int qrow_i = qb * 64 + wave * 16 + half * 8 + r;
          const int kcol   = kb * 64 + j * 16 + nI;
          if (kcol > qrow_i) x = -1.0e30f;
        }
        sv[j][r] = x;
      }

    // row max over 64 keys: per-lane partial then 16-lane xor reduction
    float pm[8];
#pragma unroll
    for (int r = 0; r < 8; ++r)
      pm[r] = fmaxf(fmaxf(sv[0][r], sv[1][r]), fmaxf(sv[2][r], sv[3][r]));
    for (int off = 1; off < 16; off <<= 1)
#pragma unroll
      for (int r = 0; r < 8; ++r)
        pm[r] = fmaxf(pm[r], __shfl_xor(pm[r], off, 32));

    float alpha[8], mn[8];
#pragma unroll
    for (int r = 0; r < 8; ++r) {
      mn[r]    = fmaxf(mrow[r], pm[r]);
      alpha[r] = __expf(mrow[r] - mn[r]);
      mrow[r]  = mn[r];
    }

    // P = exp(S - m): row sums + spill to LDS in A-matrix order
    float rs[8];
#pragma unroll
    for (int r = 0; r < 8; ++r) rs[r] = 0.0f;
#pragma unroll
    for (int j = 0; j < 4; ++j)
#pragma unroll
      for (int r = 0; r < 8; ++r) {
        const float p = __expf(sv[j][r] - mn[r]);
        rs[r] += p;
        Ps[wave][(half * 8 + r) * 72 + j * 16 + nI] = (bf16)p;
      }
    for (int off = 1; off < 16; off <<= 1)
#pragma unroll
      for (int r = 0; r < 8; ++r) rs[r] += __shfl_xor(rs[r], off, 32);
#pragma unroll
    for (int r = 0; r < 8; ++r) lrow[r] = lrow[r] * alpha[r] + rs[r];

    // rescale O accumulators by alpha (C row = r + 8*half matches)
#pragma unroll
    for (int j = 0; j < 8; ++j)
#pragma unroll
      for (int r = 0; r < 8; ++r) oacc[j][r] *= alpha[r];

    // O += P * V   (same-wave LDS ops are in-order: store->load safe)
    v16bf ap[2];
#pragma unroll
    for (int d = 0; d < 2; ++d)
      ap[d] = load_frag_a(&Ps[wave][d * 32], 72);
#pragma unroll
    for (int j = 0; j < 8; ++j)
#pragma unroll
      for (int d = 0; d < 2; ++d)
        oacc[j] = __builtin_amdgcn_wmma_f32_16x16x32_bf16(
            false, ap[d], false, load_frag_b(Vs + (j * 16) * 68 + d * 32, 68),
            (short)0, oacc[j], false, false);
  }

  // normalize and write bf16 attn output (head-concatenated [L][EMBED])
  float inv[8];
#pragma unroll
  for (int r = 0; r < 8; ++r) inv[r] = 1.0f / lrow[r];
#pragma unroll
  for (int j = 0; j < 8; ++j)
#pragma unroll
    for (int r = 0; r < 8; ++r) {
      const size_t row = (size_t)(qb * 64 + wave * 16 + half * 8 + r);
      const size_t col = (size_t)(h * DH + j * 16 + nI);
      O[row * (size_t)EMBED + col] = (bf16)(oacc[j][r] * inv[r]);
    }
}

// ---------------------------------------------------------------------------
extern "C" void kernel_launch(void* const* d_in, const int* in_sizes, int n_in,
                              void* d_out, int out_size, void* d_ws, size_t ws_size,
                              hipStream_t stream) {
  const float* x  = (const float*)d_in[0];
  const float* Wq = (const float*)d_in[1];
  const float* bq = (const float*)d_in[2];
  const float* Wk = (const float*)d_in[3];
  const float* bk = (const float*)d_in[4];
  const float* Wv = (const float*)d_in[5];
  const float* bv = (const float*)d_in[6];
  const float* Wo = (const float*)d_in[7];
  const float* bo = (const float*)d_in[8];
  const int* is_causal = (const int*)d_in[9];

  const size_t plane = (size_t)SEQ * EMBED;   // 8M elements
  bf16* qws = (bf16*)d_ws;
  bf16* kws = qws + plane;
  bf16* vws = kws + plane;
  bf16* aws = vws + plane;                    // 64 MB of workspace total

  dim3 gproj(EMBED / 128, SEQ / 128);         // (16, 32)
  gemm_xwt_wmma<false, true><<<gproj, 256, 0, stream>>>(
      x, nullptr, Wq, bq, nullptr, qws, SEQ, EMBED, EMBED);
  gemm_xwt_wmma<false, true><<<gproj, 256, 0, stream>>>(
      x, nullptr, Wk, bk, nullptr, kws, SEQ, EMBED, EMBED);
  gemm_xwt_wmma<false, true><<<gproj, 256, 0, stream>>>(
      x, nullptr, Wv, bv, nullptr, vws, SEQ, EMBED, EMBED);

  dim3 gattn(SEQ / 64, HEADS);                // (64, 16)
  flash_attn_wmma<<<gattn, 128, 0, stream>>>(qws, kws, vws, aws, is_causal);

  gemm_xwt_wmma<true, false><<<gproj, 256, 0, stream>>>(
      nullptr, aws, Wo, bo, (float*)d_out, nullptr, SEQ, EMBED, EMBED);
}
